// Router_2723009265964
// MI455X (gfx1250) — matrix operand order repaired and checked by reading, not compiled, not run on hardware
//
#include <hip/hip_runtime.h>
#include <hip/hip_bf16.h>
#include <math.h>

typedef __attribute__((ext_vector_type(16))) __bf16 v16bf;
typedef __attribute__((ext_vector_type(8)))  float  v8f;
typedef __attribute__((ext_vector_type(4)))  float  vf4;   // clang vector (nontemporal-ok)

// Problem constants (match reference)
constexpr int Dn    = 2048;           // hidden dim (K of GEMM)
constexpr int En    = 64;             // experts (N of GEMM)
constexpr int BTn   = 8192;           // B*T tokens (M of GEMM)
constexpr int KT    = Dn / 32;        // 64 k-tiles of 32
constexpr int NT    = En / 16;        // 4 n-tiles of 16
constexpr int FRAG  = 512;            // 32x16 bf16 elements per B fragment
constexpr int WELEM = Dn * En;        // 131072 elements per weight matrix
constexpr int TOK_PER_BLOCK = 128;    // 8 waves x 16 tokens
constexpr int LDS_STRIDE = 66;        // 64 + 2 pad (kills half-wave bank conflicts)

// ---------------------------------------------------------------------------
// Prep: repack f32 weights [D][E] into bf16 WMMA B-fragments.
// Fragment f = kt*NT + nt holds the 32x16 tile; within a fragment lane l's 16
// values are contiguous (lane<16 -> col n=l, K=0..7,16..23 ; lane>=16 -> col n=l-16,
// K=8..15,24..31), so the main kernel loads each fragment with two b128 loads.
// ---------------------------------------------------------------------------
__global__ void prep_weights_kernel(const float* __restrict__ wg,
                                    const float* __restrict__ wn,
                                    __bf16* __restrict__ outg,
                                    __bf16* __restrict__ outn) {
    int idx = blockIdx.x * blockDim.x + threadIdx.x;
    if (idx >= WELEM) return;
    int f  = idx >> 9;       // fragment id
    int r  = idx & 511;      // element within fragment
    int l  = r >> 4;         // lane 0..31
    int j  = r & 15;         // element within lane
    int kt = f >> 2;         // f / NT
    int nt = f & 3;          // f % NT
    int k  = kt * 32 + ((l >= 16) ? 8 : 0) + ((j < 8) ? j : j + 8);
    int n  = nt * 16 + (l & 15);
    outg[idx] = (__bf16)wg[k * En + n];
    outn[idx] = (__bf16)wn[k * En + n];
}

// Per-k-step operand bundle (kept entirely in VGPRs; two live copies = double buffer)
struct Frags {
    vf4   f0, f1, f2, f3;                 // raw f32 A data
    v16bf g0, g1, g2, g3;                 // gate B fragments
    v16bf n0, n1, n2, n3;                 // noise B fragments
};

// ---------------------------------------------------------------------------
// Main fused router kernel: dual GEMM (gate + noise) via bf16 WMMA with
// explicit double-buffered software pipeline, fused softplus-noise epilogue,
// per-token top-2, 2-way softmax, scatter + one-hot mask.
// ---------------------------------------------------------------------------
__global__ void __launch_bounds__(256)
router_main_kernel(const float* __restrict__ x,
                   const float* __restrict__ eps,
                   const __bf16* __restrict__ wsG,
                   const __bf16* __restrict__ wsN,
                   float* __restrict__ probs,
                   int* __restrict__ topi,
                   int* __restrict__ mask) {
    __shared__ float sLog[TOK_PER_BLOCK * LDS_STRIDE];
    __shared__ int   sIdx[TOK_PER_BLOCK * 2];
    __shared__ float sP[TOK_PER_BLOCK * 2];

    const int tid  = threadIdx.x;
    const int wave = tid >> 5;
    const int lane = tid & 31;
    const int l15  = lane & 15;
    const int tokBase = blockIdx.x * TOK_PER_BLOCK;
    const int m0 = tokBase + wave * 16;

    v8f cg[NT], cn[NT];
#pragma unroll
    for (int nt = 0; nt < NT; ++nt) {
        cg[nt] = (v8f){0.f, 0.f, 0.f, 0.f, 0.f, 0.f, 0.f, 0.f};
        cn[nt] = (v8f){0.f, 0.f, 0.f, 0.f, 0.f, 0.f, 0.f, 0.f};
    }

    // A source: lane<16 reads row m0+lane at K-halves {0..7,16..23},
    // lane>=16 reads row m0+lane-16 at K-halves {8..15,24..31}.
    const int arow = m0 + l15;
    const float* xrow = x + (size_t)arow * Dn + ((lane < 16) ? 0 : 8);
    const __bf16* bgBase = wsG + lane * 16;
    const __bf16* bnBase = wsN + lane * 16;

    // Issue all 20 loads (4x b128 NT for A, 16x b128 for B) for one k-step.
    auto loadStep = [&](int kt, Frags& F) {
        const float* pa = xrow + kt * 32;
        F.f0 = __builtin_nontemporal_load((const vf4*)(pa));
        F.f1 = __builtin_nontemporal_load((const vf4*)(pa + 4));
        F.f2 = __builtin_nontemporal_load((const vf4*)(pa + 16));
        F.f3 = __builtin_nontemporal_load((const vf4*)(pa + 20));
        const __bf16* bg = bgBase + (size_t)kt * (NT * FRAG);
        const __bf16* bn = bnBase + (size_t)kt * (NT * FRAG);
        F.g0 = *(const v16bf*)(bg + 0 * FRAG);
        F.g1 = *(const v16bf*)(bg + 1 * FRAG);
        F.g2 = *(const v16bf*)(bg + 2 * FRAG);
        F.g3 = *(const v16bf*)(bg + 3 * FRAG);
        F.n0 = *(const v16bf*)(bn + 0 * FRAG);
        F.n1 = *(const v16bf*)(bn + 1 * FRAG);
        F.n2 = *(const v16bf*)(bn + 2 * FRAG);
        F.n3 = *(const v16bf*)(bn + 3 * FRAG);
    };

    // Convert A to bf16 and run the 8 WMMAs of one k-step.
    auto computeStep = [&](const Frags& F) {
        v16bf a;
        a[0]  = (__bf16)F.f0[0]; a[1]  = (__bf16)F.f0[1]; a[2]  = (__bf16)F.f0[2]; a[3]  = (__bf16)F.f0[3];
        a[4]  = (__bf16)F.f1[0]; a[5]  = (__bf16)F.f1[1]; a[6]  = (__bf16)F.f1[2]; a[7]  = (__bf16)F.f1[3];
        a[8]  = (__bf16)F.f2[0]; a[9]  = (__bf16)F.f2[1]; a[10] = (__bf16)F.f2[2]; a[11] = (__bf16)F.f2[3];
        a[12] = (__bf16)F.f3[0]; a[13] = (__bf16)F.f3[1]; a[14] = (__bf16)F.f3[2]; a[15] = (__bf16)F.f3[3];
        cg[0] = __builtin_amdgcn_wmma_f32_16x16x32_bf16(false, a, false, F.g0, (short)0, cg[0], false, false);
        cn[0] = __builtin_amdgcn_wmma_f32_16x16x32_bf16(false, a, false, F.n0, (short)0, cn[0], false, false);
        cg[1] = __builtin_amdgcn_wmma_f32_16x16x32_bf16(false, a, false, F.g1, (short)0, cg[1], false, false);
        cn[1] = __builtin_amdgcn_wmma_f32_16x16x32_bf16(false, a, false, F.n1, (short)0, cn[1], false, false);
        cg[2] = __builtin_amdgcn_wmma_f32_16x16x32_bf16(false, a, false, F.g2, (short)0, cg[2], false, false);
        cn[2] = __builtin_amdgcn_wmma_f32_16x16x32_bf16(false, a, false, F.n2, (short)0, cn[2], false, false);
        cg[3] = __builtin_amdgcn_wmma_f32_16x16x32_bf16(false, a, false, F.g3, (short)0, cg[3], false, false);
        cn[3] = __builtin_amdgcn_wmma_f32_16x16x32_bf16(false, a, false, F.n3, (short)0, cn[3], false, false);
    };

    // Double-buffered pipeline: loads for step k+1 are in flight while the
    // WMMAs of step k issue, so loadcnt waits land behind real work.
    Frags Fa, Fb;
    loadStep(0, Fa);
#pragma unroll 1
    for (int kt = 0; kt < KT - 2; kt += 2) {
        loadStep(kt + 1, Fb);
        computeStep(Fa);
        loadStep(kt + 2, Fa);
        computeStep(Fb);
    }
    loadStep(KT - 1, Fb);   // kt = KT-2 tail
    computeStep(Fa);
    computeStep(Fb);

    // Fused epilogue at C-store time: logit = gate + softplus(noise)*eps.
    // C layout: VGPR r, lanes 0-15 -> M=r, N=lane ; lanes 16-31 -> M=r+8.
    const int rowOff = (lane < 16) ? 0 : 8;
#pragma unroll
    for (int nt = 0; nt < NT; ++nt) {
        const int n = nt * 16 + l15;
#pragma unroll
        for (int r = 0; r < 8; ++r) {
            const int tl = wave * 16 + r + rowOff;
            float g = cg[nt][r];
            float z = cn[nt][r];
            float ep = __builtin_nontemporal_load(
                eps + (size_t)(tokBase + tl) * En + n);
            float sp = (z > 20.f) ? z : log1pf(__expf(z));  // softplus
            sLog[tl * LDS_STRIDE + n] = g + sp * ep;
        }
    }
    __syncthreads();

    // Per-token top-2 + softmax over the two values (ties keep lowest index,
    // matching jax.lax.top_k).
    if (tid < TOK_PER_BLOCK) {
        const float* row = sLog + tid * LDS_STRIDE;
        float m1 = -3.402823466e38f, m2 = -3.402823466e38f;
        int i1 = 0, i2 = 0;
        for (int e = 0; e < En; ++e) {
            float v = row[e];
            if (v > m1) { m2 = m1; i2 = i1; m1 = v; i1 = e; }
            else if (v > m2) { m2 = v; i2 = e; }
        }
        float e2 = __expf(m2 - m1);
        float inv = 1.f / (1.f + e2);
        sIdx[tid * 2]     = i1;
        sIdx[tid * 2 + 1] = i2;
        sP[tid * 2]       = inv;
        sP[tid * 2 + 1]   = e2 * inv;
    }
    __syncthreads();

    // Coalesced output writes (harness poisons d_out: write every element).
    // All outputs are write-once / never re-read -> non-temporal stores.
    // router_probs [BT, E]
    for (int i = tid; i < TOK_PER_BLOCK * En; i += 256) {
        int t = i >> 6, e = i & 63;
        int i1 = sIdx[t * 2], i2 = sIdx[t * 2 + 1];
        float v = (e == i1) ? sP[t * 2] : (e == i2) ? sP[t * 2 + 1] : 0.f;
        __builtin_nontemporal_store(v, probs + (size_t)(tokBase + t) * En + e);
    }
    // top_k_indices [BT, 2]
    if (tid < TOK_PER_BLOCK * 2) {
        int t = tid >> 1, k = tid & 1;
        __builtin_nontemporal_store(sIdx[t * 2 + k],
                                    topi + (size_t)(tokBase + t) * 2 + k);
    }
    // exp_mask [2, BT, E]
    for (int i = tid; i < 2 * TOK_PER_BLOCK * En; i += 256) {
        int k = i >> 13;            // i / (128*64)
        int rem = i & 8191;
        int t = rem >> 6, e = rem & 63;
        int v = (sIdx[t * 2 + k] == e) ? 1 : 0;
        __builtin_nontemporal_store(
            v, mask + (size_t)k * BTn * En + (size_t)(tokBase + t) * En + e);
    }
}

// ---------------------------------------------------------------------------
extern "C" void kernel_launch(void* const* d_in, const int* in_sizes, int n_in,
                              void* d_out, int out_size, void* d_ws, size_t ws_size,
                              hipStream_t stream) {
    const float* x   = (const float*)d_in[0];   // [4,2048,2048]
    const float* eps = (const float*)d_in[1];   // [4,2048,64]
    const float* wg  = (const float*)d_in[2];   // [2048,64]
    const float* wn  = (const float*)d_in[3];   // [2048,64]

    __bf16* wsG = (__bf16*)d_ws;                // 256 KB
    __bf16* wsN = wsG + WELEM;                  // 256 KB

    float* probs = (float*)d_out;                       // 524288 f32
    int*   topi  = (int*)d_out + (size_t)BTn * En;      // 16384 i32
    int*   mask  = topi + (size_t)BTn * 2;              // 1048576 i32

    prep_weights_kernel<<<(WELEM + 255) / 256, 256, 0, stream>>>(wg, wn, wsG, wsN);
    router_main_kernel<<<BTn / TOK_PER_BLOCK, 256, 0, stream>>>(
        x, eps, wsG, wsN, probs, topi, mask);
}
